// HyperNN_27066883900215
// MI455X (gfx1250) — matrix operand order repaired and checked by reading, not compiled, not run on hardware
//
#include <hip/hip_runtime.h>

// ---------------------------------------------------------------------------
// HyperNN batched MLP:  out = relu(relu(z@W1+b1)@W2+b2)@W3+b3
//   z: [1048576, 32] f32, W1:[32,128], W2:[128,128], W3:[128,64], out f32.
// Memory-bound (~402 MB @ 23.3 TB/s => ~17us floor). f16 WMMA
// (v_wmma_f32_16x16x32_f16) with f32 accumulation keeps compute at the same
// order as the memory floor. Each wave processes TWO 16-row tiles (M=32) so
// every B-fragment LDS load feeds two WMMAs and there are two independent
// dependency chains to hide WMMA latency / DS waits / hazard NOPs.
// ---------------------------------------------------------------------------

typedef _Float16 half_t;
typedef __attribute__((ext_vector_type(16))) _Float16 v16h;
typedef __attribute__((ext_vector_type(8)))  _Float16 v8h;
typedef __attribute__((ext_vector_type(8)))  float    v8f;

#define Z_NUM 1048576
#define Z_DIM 32
#define HID   128
#define OUT_D 64

// LDS strides in halfs; +8 halfs (16B) pad => 4-bank rotation per row
#define S1 136   // W1 rows (128 cols + 8 pad)
#define S2 136   // W2 rows
#define S3 72    // W3 rows (64 cols + 8 pad)
#define SH 136   // activation buffer rows

#define WAVES   4
#define THREADS (WAVES * 32)
#define GRID    2048
#define TILES32 (Z_NUM / 32)                    // 32768 dual-tiles
#define ITERS   (TILES32 / (GRID * WAVES))      // 4

// LDS layout offsets (in halfs)
#define W1_OFF 0
#define W2_OFF (W1_OFF + 32 * S1)
#define W3_OFF (W2_OFF + HID * S2)
#define HB_OFF (W3_OFF + HID * S3)
#define LDS_HALFS (HB_OFF + WAVES * 32 * SH)    // 32-row h buffer per wave

// B fragment (32x16, f16): lane l holds col N = n0 + (l&15); elems 0..7 = K k0+8*(l>>4)+e, elems 8..15 = +16.
__device__ __forceinline__ v16h load_bfrag(const half_t* __restrict__ w,
                                           int stride, int k0, int n0, int lane) {
    const half_t* p = w + (size_t)(k0 + 8 * (lane >> 4)) * stride + n0 + (lane & 15);
    v16h b;
#pragma unroll
    for (int i = 0; i < 8; ++i) { b[i] = p[(size_t)i * stride]; b[i + 8] = p[(size_t)(16 + i) * stride]; }
    return b;
}

// A fragment (16x32, f16) from LDS activation buffer (row-major, stride SH).
// Lane l: row = l&15; elems 0..7 = K kt*32+8*(l>>4)+e, elems 8..15 = +16.
__device__ __forceinline__ v16h load_afrag_lds(const half_t* __restrict__ h,
                                               int kt, int lane) {
    const int row = lane & 15, hi4 = lane >> 4;
    const half_t* p = h + (size_t)row * SH + kt * 32 + 8 * hi4;
    v8h lo = *(const v8h*)p;          // K = kt*32 + 8*hi4 + 0..7
    v8h hi = *(const v8h*)(p + 16);   // K = kt*32 + 16 + 8*hi4 + 0..7
    v16h a;
#pragma unroll
    for (int i = 0; i < 8; ++i) { a[i] = lo[i]; a[i + 8] = hi[i]; }
    return a;
}

// D (16x16 f32 accum) -> relu(+bias) -> f16 activation buffer.
// Lane l, elem r holds (M = r + 8*(l>>4), N = n0 + (l&15)).
__device__ __forceinline__ void store_h(half_t* __restrict__ h, v8f c,
                                        float bias, int n0, int lane) {
    const int col = n0 + (lane & 15);
    const int rbase = (lane >> 4) * 8;
#pragma unroll
    for (int r = 0; r < 8; ++r) {
        float v = c[r] + bias;
        v = v > 0.0f ? v : 0.0f;
        h[(size_t)(rbase + r) * SH + col] = (half_t)v;
    }
}

// A fragment straight from a global f32 row block (16 rows starting at row0).
__device__ __forceinline__ v16h load_afrag_z(const float* __restrict__ z,
                                             int row0, int lane) {
    const int row16 = lane & 15, hi4 = lane >> 4;
    const float* zp = z + (size_t)(row0 + row16) * Z_DIM;
    v16h a;
#pragma unroll
    for (int e = 0; e < 8; ++e) {
        a[e]     = (half_t)zp[8 * hi4 + e];        // K = 8*hi4 + e
        a[e + 8] = (half_t)zp[16 + 8 * hi4 + e];   // K = 16 + 8*hi4 + e
    }
    return a;
}

__global__ __launch_bounds__(THREADS)
void HyperNN_mlp_wmma(const float* __restrict__ z,
                      const float* __restrict__ W1, const float* __restrict__ b1,
                      const float* __restrict__ W2, const float* __restrict__ b2,
                      const float* __restrict__ W3, const float* __restrict__ b3,
                      float* __restrict__ out) {
    extern __shared__ half_t lds[];
    half_t* w1 = lds + W1_OFF;
    half_t* w2 = lds + W2_OFF;
    half_t* w3 = lds + W3_OFF;

    const int tid = threadIdx.x;

    // --- convert weights f32 -> f16 into padded LDS (once per block) ---
    for (int i = tid; i < 32 * HID; i += THREADS)
        w1[(size_t)(i >> 7) * S1 + (i & 127)] = (half_t)W1[i];
    for (int i = tid; i < HID * HID; i += THREADS)
        w2[(size_t)(i >> 7) * S2 + (i & 127)] = (half_t)W2[i];
    for (int i = tid; i < HID * OUT_D; i += THREADS)
        w3[(size_t)(i >> 6) * S3 + (i & 63)] = (half_t)W3[i];
    __syncthreads();

    const int lane  = tid & 31;
    const int wave  = tid >> 5;
    const int row16 = lane & 15;
    const int hi4   = lane >> 4;
    half_t* hb0 = lds + HB_OFF + (size_t)wave * 32 * SH;  // private per wave
    half_t* hb1 = hb0 + (size_t)16 * SH;

    for (int it = 0; it < ITERS; ++it) {
        const int tile = (blockIdx.x * WAVES + wave) + it * (GRID * WAVES);
        const int row0 = tile * 32;
        if (row0 >= Z_NUM) break;

        // prefetch next iteration's z rows (global_prefetch_b8)
        if (it + 1 < ITERS) {
            const size_t nrow = (size_t)(row0 + row16) * Z_DIM
                              + (size_t)(GRID * WAVES) * 32 * Z_DIM;
            __builtin_prefetch(z + nrow, 0, 0);
        }

        // ---- A1 fragments for both 16-row tiles (f32 -> f16) ----
        v16h a1_0 = load_afrag_z(z, row0,      lane);
        v16h a1_1 = load_afrag_z(z, row0 + 16, lane);

        // ---- layer 1: h = relu(z @ W1 + b1), K=32 ----
#pragma unroll
        for (int nt = 0; nt < 8; ++nt) {
            v16h b = load_bfrag(w1, S1, 0, nt * 16, lane);
            v8f c0 = {}, c1 = {};
            c0 = __builtin_amdgcn_wmma_f32_16x16x32_f16(false, a1_0, false, b,
                                                        (short)0, c0, false, false);
            c1 = __builtin_amdgcn_wmma_f32_16x16x32_f16(false, a1_1, false, b,
                                                        (short)0, c1, false, false);
            const float bias = b1[nt * 16 + row16];
            store_h(hb0, c0, bias, nt * 16, lane);
            store_h(hb1, c1, bias, nt * 16, lane);
        }

        // ---- gather layer-2 A fragments (2 tiles x 4 K-tiles) ----
        v16h a2[2][4];
#pragma unroll
        for (int kt = 0; kt < 4; ++kt) {
            a2[0][kt] = load_afrag_lds(hb0, kt, lane);
            a2[1][kt] = load_afrag_lds(hb1, kt, lane);
        }

        // ---- layer 2: h = relu(h @ W2 + b2), K=128 (overwrites hb) ----
#pragma unroll
        for (int nt = 0; nt < 8; ++nt) {
            v8f c0 = {}, c1 = {};
#pragma unroll
            for (int kt = 0; kt < 4; ++kt) {
                v16h b = load_bfrag(w2, S2, kt * 32, nt * 16, lane);
                c0 = __builtin_amdgcn_wmma_f32_16x16x32_f16(false, a2[0][kt], false, b,
                                                            (short)0, c0, false, false);
                c1 = __builtin_amdgcn_wmma_f32_16x16x32_f16(false, a2[1][kt], false, b,
                                                            (short)0, c1, false, false);
            }
            const float bias = b2[nt * 16 + row16];
            store_h(hb0, c0, bias, nt * 16, lane);
            store_h(hb1, c1, bias, nt * 16, lane);
        }

        // ---- gather layer-3 A fragments ----
        v16h a3[2][4];
#pragma unroll
        for (int kt = 0; kt < 4; ++kt) {
            a3[0][kt] = load_afrag_lds(hb0, kt, lane);
            a3[1][kt] = load_afrag_lds(hb1, kt, lane);
        }

        // ---- layer 3: out = h @ W3 + b3, K=128, 64 cols, f32 to global ----
#pragma unroll
        for (int nt = 0; nt < 4; ++nt) {
            v8f c0 = {}, c1 = {};
#pragma unroll
            for (int kt = 0; kt < 4; ++kt) {
                v16h b = load_bfrag(w3, S3, kt * 32, nt * 16, lane);
                c0 = __builtin_amdgcn_wmma_f32_16x16x32_f16(false, a3[0][kt], false, b,
                                                            (short)0, c0, false, false);
                c1 = __builtin_amdgcn_wmma_f32_16x16x32_f16(false, a3[1][kt], false, b,
                                                            (short)0, c1, false, false);
            }
            const float bias = b3[nt * 16 + row16];
            float* op0 = out + (size_t)(row0 + hi4 * 8) * OUT_D + nt * 16 + row16;
            float* op1 = op0 + (size_t)16 * OUT_D;
#pragma unroll
            for (int r = 0; r < 8; ++r) {
                op0[(size_t)r * OUT_D] = c0[r] + bias;
                op1[(size_t)r * OUT_D] = c1[r] + bias;
            }
        }
    }
}

extern "C" void kernel_launch(void* const* d_in, const int* in_sizes, int n_in,
                              void* d_out, int out_size, void* d_ws, size_t ws_size,
                              hipStream_t stream) {
    (void)in_sizes; (void)n_in; (void)out_size; (void)d_ws; (void)ws_size;
    const float* z  = (const float*)d_in[0];
    const float* W1 = (const float*)d_in[1];
    const float* b1 = (const float*)d_in[2];
    const float* W2 = (const float*)d_in[3];
    const float* b2 = (const float*)d_in[4];
    const float* W3 = (const float*)d_in[5];
    const float* b3 = (const float*)d_in[6];
    float* out = (float*)d_out;

    const size_t shmem = (size_t)LDS_HALFS * sizeof(half_t);   // ~94.5 KB
    HyperNN_mlp_wmma<<<GRID, THREADS, shmem, stream>>>(z, W1, b1, W2, b2, W3, b3, out);
}
